// GCNNet_59811714564605
// MI455X (gfx1250) — compile-verified, matrix operand order checked by reading, not run on hardware
//
#include <hip/hip_runtime.h>
#include <hip/hip_bf16.h>
#include <cstdint>
#include <cstddef>

#ifndef __has_builtin
#define __has_builtin(x) 0
#endif

typedef float  v2f   __attribute__((ext_vector_type(2)));
typedef float  v8f   __attribute__((ext_vector_type(8)));
typedef __bf16 v16bf __attribute__((ext_vector_type(16)));
typedef unsigned int u32x4 __attribute__((ext_vector_type(4)));
typedef int          i32x4 __attribute__((ext_vector_type(4)));
typedef int          i32x8 __attribute__((ext_vector_type(8)));

#define NNODES 20000
#define NEDGES 320000
#define NGRAPH 512

#define KC 64            // K-chunk staged in LDS per iteration
#define SA_STRIDE 72     // padded LDS row stride for A chunk (floats)

#if defined(__HIP_DEVICE_COMPILE__) && __has_builtin(__builtin_amdgcn_wmma_f32_16x16x4_f32)
#define WMMA_HAS_F32X4 1
#else
#define WMMA_HAS_F32X4 0
#endif

#if defined(__HIP_DEVICE_COMPILE__) && __has_builtin(__builtin_amdgcn_tensor_load_to_lds)
#define HAVE_TDM 1
#else
#define HAVE_TDM 0
#endif

#if HAVE_TDM
// Issue one TDM 2D tile load: rows x 16 f32 tile, row stride = stride0 elements,
// zero-fill beyond rowsAvail (tensor_dim1). D# packed per CDNA5 ISA §8.3/8.4.
__device__ __forceinline__ void tdm_load_b_tile(unsigned lds_off, const float* gptr,
                                                unsigned rowsAvail, unsigned stride0) {
  const unsigned long long ga = (unsigned long long)(size_t)(const void*)gptr;
  u32x4 g0;
  g0.x = 1u;                                                  // count=1 (valid), user mode
  g0.y = lds_off;                                             // lds_addr (bytes)
  g0.z = (unsigned)(ga & 0xFFFFFFFFull);                      // global_addr[31:0]
  g0.w = (unsigned)((ga >> 32) & 0x1FFFFFFull) | (2u << 30);  // global_addr[56:32] | type=2
  const unsigned td0 = 16u;                                   // tensor_dim0 (row length)
  i32x8 g1;
  g1[0] = (int)(2u << 16);                                    // workgroup_mask=0, data_size=2 (4B)
  g1[1] = (int)((td0 & 0xFFFFu) << 16);                       // tensor_dim0 lo16 (abar addr=0)
  g1[2] = (int)((td0 >> 16) | ((rowsAvail & 0xFFFFu) << 16)); // dim0 hi16 | tensor_dim1 lo16
  g1[3] = (int)((rowsAvail >> 16) | (16u << 16));             // dim1 hi16 | tile_dim0=16
  g1[4] = (int)(KC);                                          // tile_dim1=KC rows, tile_dim2=0
  g1[5] = (int)(stride0);                                     // tensor_dim0_stride lo32
  g1[6] = 0;                                                  // stride0 hi16 | dim1_stride lo=0
  g1[7] = 0;
  const i32x4 gz = {0, 0, 0, 0};
#if defined(__clang_major__) && __clang_major__ >= 23
  const i32x8 gz8 = {0, 0, 0, 0, 0, 0, 0, 0};
  __builtin_amdgcn_tensor_load_to_lds(g0, g1, gz, gz, gz8, 0);
#else
  __builtin_amdgcn_tensor_load_to_lds(g0, g1, gz, gz, 0);
#endif
}
#endif

// ---------------------------------------------------------------------------
// GEMM: C[M,Nc] = A[M,K] @ B[K,Nc] (row-major fp32).
// Block = 256 threads = 8 waves; block tile = 128 rows x 16 cols.
// Wave w owns the 16x16 tile at row (blockIdx.y*8 + w)*16 (clamped; duplicate
// waves recompute identical data -> benign identical stores).
// K is consumed in LDS-staged chunks of KC with zero-fill, so the WMMA loop is
// completely branch-free (EXEC all-ones as WMMA requires).
// ---------------------------------------------------------------------------
__global__ __launch_bounds__(256) void gemm_wmma(const float* __restrict__ A,
                                                 const float* __restrict__ B,
                                                 float* __restrict__ C,
                                                 int M, int K, int Nc) {
  __shared__ float sA[128 * SA_STRIDE];
  __shared__ float sB[KC * 16];

  const int lane = threadIdx.x & 31;
  const int wave = threadIdx.x >> 5;
  const int half = lane >> 4;          // 0: lanes 0-15, 1: lanes 16-31
  const int l16  = lane & 15;

  const int col0 = blockIdx.x << 4;
  const int rowBase = blockIdx.y << 7;                 // 128 rows per block
  const int tileRow = min((rowBase >> 4) + wave, (M >> 4) - 1);
  const int rLoc = (tileRow << 4) - rowBase;           // this wave's rows inside sA

  v8f acc = {};
#if HAVE_TDM
  const unsigned sB_off = (unsigned)(size_t)(void*)sB; // LDS byte offset
#endif

  for (int k = 0; k < K; k += KC) {
    __syncthreads();   // previous chunk fully consumed before restaging

    // ---- stage A chunk: 128 rows x KC, zero-filled past K
    for (int i = threadIdx.x; i < 128 * KC; i += 256) {
      const int r = i >> 6, kk = i & (KC - 1);
      const int gr = min(rowBase + r, M - 1);
      const int gk = k + kk;
      sA[r * SA_STRIDE + kk] = (gk < K) ? A[(size_t)gr * K + gk] : 0.f;
    }

    // ---- stage B chunk (KC x 16 strip)
#if HAVE_TDM
    if (wave == 0) {   // TDM is a per-wave DMA; one wave issues for the block
      tdm_load_b_tile(sB_off, B + (size_t)k * Nc + col0,
                      (unsigned)(K - k), (unsigned)Nc);
      __builtin_amdgcn_s_wait_tensorcnt(0);
    }
#else
    for (int i = threadIdx.x; i < KC * 16; i += 256) {
      const int kk = i >> 4, c = i & 15;
      const int gk = k + kk;
      sB[kk * 16 + c] = (gk < K) ? B[(size_t)gk * Nc + col0 + c] : 0.f;
    }
#endif

    // ---- prefetch next A chunk into caches (global_prefetch_b8)
    if (k + KC < K) {
      const int pr = min(rowBase + (int)(threadIdx.x >> 1), M - 1);
      const int pk = k + KC + ((threadIdx.x & 1) << 5);
      __builtin_prefetch(&A[(size_t)pr * K + pk], 0, 1);
    }

    __syncthreads();   // staged data visible to all waves

    // ---- branch-free MAC loop over the chunk
    const float* aBase = &sA[(size_t)(rLoc + l16) * SA_STRIDE];
#if WMMA_HAS_F32X4
#pragma unroll
    for (int s = 0; s < KC / 4; ++s) {
      const int kb = (s << 2) + (half << 1);           // even -> 8B-aligned
      const v2f a = *(const v2f*)(aBase + kb);
      v2f b;
      b.x = sB[kb * 16 + l16];
      b.y = sB[(kb + 1) * 16 + l16];
      acc = __builtin_amdgcn_wmma_f32_16x16x4_f32(false, a, false, b,
                                                  (short)0, acc, false, false);
    }
#else
#pragma unroll
    for (int s = 0; s < KC / 32; ++s) {
      v16bf a = {}, b = {};
#pragma unroll
      for (int i = 0; i < 16; ++i) {
        const int v = i >> 1, e = i & 1;
        const int kk = (s << 5) + ((v < 4) ? (2 * v + e) : (16 + 2 * (v - 4) + e)) + (half << 3);
        a[i] = (__bf16)aBase[kk];
        b[i] = (__bf16)sB[kk * 16 + l16];
      }
      acc = __builtin_amdgcn_wmma_f32_16x16x32_bf16(false, a, false, b,
                                                    (short)0, acc, false, false);
    }
#endif
  }

  // C/D layout: VGPR j -> row (half*8 + j), col = l16
#pragma unroll
  for (int j = 0; j < 8; ++j)
    C[(size_t)((tileRow << 4) + (half << 3) + j) * Nc + col0 + l16] = acc[j];
}

// ---------------------------------------------------------------------------
// Degree / normalization
// ---------------------------------------------------------------------------
__global__ void fill_ones(float* p, int n) {
  int i = blockIdx.x * blockDim.x + threadIdx.x;
  if (i < n) p[i] = 1.0f;                      // self-loop contributes 1 to deg
}
__global__ void deg_edges(const int* __restrict__ col, float* __restrict__ deg, int e) {
  int i = blockIdx.x * blockDim.x + threadIdx.x;
  if (i < e) atomicAdd(&deg[col[i]], 1.0f);
}
__global__ void compute_dinv(const float* __restrict__ deg, float* __restrict__ dinv, int n) {
  int i = blockIdx.x * blockDim.x + threadIdx.x;
  if (i < n) dinv[i] = rsqrtf(deg[i]);         // deg >= 1 always (self-loops)
}

// ---------------------------------------------------------------------------
// GCN aggregation: agg[col] += xw[row] * dinv[row]*dinv[col]  (wave per edge)
// ---------------------------------------------------------------------------
__global__ __launch_bounds__(256) void scatter_edges(const float* __restrict__ xw,
                                                     const int* __restrict__ row,
                                                     const int* __restrict__ col,
                                                     const float* __restrict__ dinv,
                                                     float* __restrict__ agg,
                                                     int e, int d) {
  const int lane = threadIdx.x & 31;
  const int wid = (blockIdx.x * blockDim.x + threadIdx.x) >> 5;
  if (wid >= e) return;
  const int r = row[wid], c = col[wid];
  const float w = dinv[r] * dinv[c];
  const float* __restrict__ src = xw + (size_t)r * d;
  float* __restrict__ dst = agg + (size_t)c * d;
  for (int f = lane; f < d; f += 32) atomicAdd(&dst[f], src[f] * w);
}

// agg += self-loop + bias, then ReLU (in place on agg)
__global__ void selfloop_bias_relu(const float* __restrict__ xw,
                                   const float* __restrict__ dinv,
                                   const float* __restrict__ bias,
                                   float* __restrict__ h, int n, int d) {
  size_t idx = (size_t)blockIdx.x * blockDim.x + threadIdx.x;
  if (idx >= (size_t)n * d) return;
  const int nn = (int)(idx / d), f = (int)(idx % d);
  const float di = dinv[nn];
  float v = h[idx] + xw[idx] * di * di + bias[f];
  h[idx] = v > 0.f ? v : 0.f;
}

// ---------------------------------------------------------------------------
// BatchNorm (training-mode, biased variance)
// ---------------------------------------------------------------------------
__global__ __launch_bounds__(256) void bn_stats(const float* __restrict__ h,
                                                float* __restrict__ mean,
                                                float* __restrict__ rstd,
                                                int n, int d) {
  const int f = blockIdx.x;
  float s = 0.f, sq = 0.f;
  for (int i = threadIdx.x; i < n; i += blockDim.x) {
    float v = h[(size_t)i * d + f];
    s += v; sq += v * v;
  }
  __shared__ float sh[256], shq[256];
  sh[threadIdx.x] = s; shq[threadIdx.x] = sq;
  __syncthreads();
  for (int o = 128; o > 0; o >>= 1) {
    if ((int)threadIdx.x < o) { sh[threadIdx.x] += sh[threadIdx.x + o];
                                shq[threadIdx.x] += shq[threadIdx.x + o]; }
    __syncthreads();
  }
  if (threadIdx.x == 0) {
    float m = sh[0] / n;
    float var = shq[0] / n - m * m;
    mean[f] = m;
    rstd[f] = rsqrtf(var + 1e-5f);
  }
}
__global__ void bn_apply(float* __restrict__ h, const float* __restrict__ mean,
                         const float* __restrict__ rstd, const float* __restrict__ g,
                         const float* __restrict__ be, int n, int d) {
  size_t idx = (size_t)blockIdx.x * blockDim.x + threadIdx.x;
  if (idx >= (size_t)n * d) return;
  const int f = (int)(idx % d);
  h[idx] = g[f] * (h[idx] - mean[f]) * rstd[f] + be[f];
}

// ---------------------------------------------------------------------------
// Global attention pooling
// ---------------------------------------------------------------------------
__device__ __forceinline__ unsigned f2ord(float f) {
  unsigned u = __float_as_uint(f);
  return (u & 0x80000000u) ? ~u : (u | 0x80000000u);
}
__device__ __forceinline__ float ord2f(unsigned k) {
  return (k & 0x80000000u) ? __uint_as_float(k ^ 0x80000000u) : __uint_as_float(~k);
}

__global__ __launch_bounds__(256) void gate_kernel(const float* __restrict__ h,
                                                   const float* __restrict__ Wg,
                                                   const float* __restrict__ bg,
                                                   float* __restrict__ gate, int n, int d) {
  const int lane = threadIdx.x & 31;
  const int wid = (blockIdx.x * blockDim.x + threadIdx.x) >> 5;
  if (wid >= n) return;
  const float* __restrict__ hr = h + (size_t)wid * d;
  float s = 0.f;
  for (int f = lane; f < d; f += 32) s += hr[f] * Wg[f];
  for (int o = 16; o > 0; o >>= 1) s += __shfl_down(s, o, 32);
  if (lane == 0) gate[wid] = s + bg[0];
}
__global__ void gmax_kernel(const float* __restrict__ gate, const int* __restrict__ batch,
                            unsigned* __restrict__ gmax, int n) {
  int i = blockIdx.x * blockDim.x + threadIdx.x;
  if (i < n) atomicMax(&gmax[batch[i]], f2ord(gate[i]));
}
__global__ void expsum_kernel(const float* __restrict__ gate, const int* __restrict__ batch,
                              const unsigned* __restrict__ gmax, float* __restrict__ e,
                              float* __restrict__ ssum, int n) {
  int i = blockIdx.x * blockDim.x + threadIdx.x;
  if (i >= n) return;
  int b = batch[i];
  float v = expf(gate[i] - ord2f(gmax[b]));
  e[i] = v;
  atomicAdd(&ssum[b], v);
}
__global__ __launch_bounds__(256) void pool_kernel(const float* __restrict__ h,
                                                   const float* __restrict__ e,
                                                   const float* __restrict__ ssum,
                                                   const int* __restrict__ batch,
                                                   float* __restrict__ pooled, int n, int d) {
  const int lane = threadIdx.x & 31;
  const int wid = (blockIdx.x * blockDim.x + threadIdx.x) >> 5;
  if (wid >= n) return;
  const int b = batch[wid];
  const float alpha = e[wid] / ssum[b];
  const float* __restrict__ hr = h + (size_t)wid * d;
  float* __restrict__ dst = pooled + (size_t)b * d;
  for (int f = lane; f < d; f += 32) atomicAdd(&dst[f], alpha * hr[f]);
}

// ---------------------------------------------------------------------------
// Tiny MLP head
// ---------------------------------------------------------------------------
__global__ void fc(const float* __restrict__ in, const float* __restrict__ W,
                   const float* __restrict__ b, float* __restrict__ out,
                   int m, int k, int nc, int relu) {
  int idx = blockIdx.x * blockDim.x + threadIdx.x;
  if (idx >= m * nc) return;
  const int r = idx / nc, j = idx % nc;
  float s = b[j];
  for (int kk = 0; kk < k; ++kk) s += in[(size_t)r * k + kk] * W[(size_t)kk * nc + j];
  out[idx] = (relu && s < 0.f) ? 0.f : s;
}

// ---------------------------------------------------------------------------
extern "C" void kernel_launch(void* const* d_in, const int* in_sizes, int n_in,
                              void* d_out, int out_size, void* d_ws, size_t ws_size,
                              hipStream_t stream) {
  (void)in_sizes; (void)n_in; (void)out_size; (void)ws_size;
  const float* x = (const float*)d_in[0];
  const int* edge = (const int*)d_in[1];
  const int* row = edge;               // edge_index[0]
  const int* col = edge + NEDGES;      // edge_index[1]
  const int* batch = (const int*)d_in[2];
  const float *W[5], *Bb[5], *Gg[5], *Be[5];
  for (int i = 0; i < 5; ++i) {
    W[i]  = (const float*)d_in[3 + 4 * i];
    Bb[i] = (const float*)d_in[4 + 4 * i];
    Gg[i] = (const float*)d_in[5 + 4 * i];
    Be[i] = (const float*)d_in[6 + 4 * i];
  }
  const float* Wg  = (const float*)d_in[23];
  const float* bg  = (const float*)d_in[24];
  const float* Wf2 = (const float*)d_in[25];
  const float* bf2 = (const float*)d_in[26];
  const float* Wf3 = (const float*)d_in[27];
  const float* bf3 = (const float*)d_in[28];
  const float* Wf4 = (const float*)d_in[29];
  const float* bf4 = (const float*)d_in[30];
  float* out = (float*)d_out;

  char* ws = (char*)d_ws;
  size_t off = 0;
  auto take = [&](size_t bytes) -> char* {
    char* p = ws + off;
    off = (off + bytes + 255) & ~(size_t)255;
    return p;
  };
  float*    buf0   = (float*)take((size_t)NNODES * 1024 * 4);
  float*    buf1   = (float*)take((size_t)NNODES * 1024 * 4);
  float*    deg    = (float*)take((size_t)NNODES * 4);
  float*    dinv   = (float*)take((size_t)NNODES * 4);
  float*    mean   = (float*)take(1024 * 4);
  float*    rstd   = (float*)take(1024 * 4);
  float*    gate   = (float*)take((size_t)NNODES * 4);
  float*    ebuf   = (float*)take((size_t)NNODES * 4);
  unsigned* gmaxb  = (unsigned*)take((size_t)NGRAPH * 4);
  float*    ssum   = (float*)take((size_t)NGRAPH * 4);
  float*    pooled = (float*)take((size_t)NGRAPH * 1024 * 4);
  float*    p2     = (float*)take((size_t)NGRAPH * 128 * 4);
  float*    p3     = (float*)take((size_t)NGRAPH * 16 * 4);

  // symmetric normalization
  fill_ones<<<(NNODES + 255) / 256, 256, 0, stream>>>(deg, NNODES);
  deg_edges<<<(NEDGES + 255) / 256, 256, 0, stream>>>(col, deg, NEDGES);
  compute_dinv<<<(NNODES + 255) / 256, 256, 0, stream>>>(deg, dinv, NNODES);

  const int dims[6] = {29, 1024, 512, 256, 512, 1024};
  const int rowTiles = NNODES / 16;                 // 1250
  const unsigned gridY = (unsigned)((rowTiles + 7) / 8);   // 8 row tiles per block
  const float* h = x;
  for (int l = 0; l < 5; ++l) {
    const int K = dims[l], D = dims[l + 1];
    dim3 grid((unsigned)(D / 16), gridY, 1);
    gemm_wmma<<<grid, 256, 0, stream>>>(h, W[l], buf1, NNODES, K, D);
    hipMemsetAsync(buf0, 0, (size_t)NNODES * D * 4, stream);
    scatter_edges<<<(NEDGES + 7) / 8, 256, 0, stream>>>(buf1, row, col, dinv, buf0, NEDGES, D);
    const size_t tot = (size_t)NNODES * D;
    selfloop_bias_relu<<<(int)((tot + 255) / 256), 256, 0, stream>>>(buf1, dinv, Bb[l], buf0, NNODES, D);
    bn_stats<<<D, 256, 0, stream>>>(buf0, mean, rstd, NNODES, D);
    bn_apply<<<(int)((tot + 255) / 256), 256, 0, stream>>>(buf0, mean, rstd, Gg[l], Be[l], NNODES, D);
    h = buf0;
  }

  // attention pooling
  gate_kernel<<<(NNODES + 7) / 8, 256, 0, stream>>>(h, Wg, bg, gate, NNODES, 1024);
  hipMemsetAsync(gmaxb, 0, (size_t)NGRAPH * 4, stream);
  hipMemsetAsync(ssum, 0, (size_t)NGRAPH * 4, stream);
  hipMemsetAsync(pooled, 0, (size_t)NGRAPH * 1024 * 4, stream);
  gmax_kernel<<<(NNODES + 255) / 256, 256, 0, stream>>>(gate, batch, gmaxb, NNODES);
  expsum_kernel<<<(NNODES + 255) / 256, 256, 0, stream>>>(gate, batch, gmaxb, ebuf, ssum, NNODES);
  pool_kernel<<<(NNODES + 7) / 8, 256, 0, stream>>>(h, ebuf, ssum, batch, pooled, NNODES, 1024);

  // MLP head
  fc<<<(NGRAPH * 128 + 255) / 256, 256, 0, stream>>>(pooled, Wf2, bf2, p2, NGRAPH, 1024, 128, 1);
  fc<<<(NGRAPH * 16 + 255) / 256, 256, 0, stream>>>(p2, Wf3, bf3, p3, NGRAPH, 128, 16, 1);
  fc<<<(NGRAPH + 255) / 256, 256, 0, stream>>>(p3, Wf4, bf4, out, NGRAPH, 16, 1, 0);
}
